// FastVecKM_58033598103723
// MI455X (gfx1250) — matrix operand order repaired and checked by reading, not compiled, not run on hardware
//
#include <hip/hip_runtime.h>
#include <hip/hip_bf16.h>

typedef __attribute__((ext_vector_type(16))) _Float16 v16h;
typedef __attribute__((ext_vector_type(8)))  float    v8f;

#define NPTS    65536
#define ENC     512
#define PP      4096
#define EAW     1024      // eA row width (2*ENC)
#define EBW     8192      // eB row width (2*P)
#define INV_R   20.0f     // 1/0.05
#define SQRT_D  22.627416997969522f  // sqrt(512)
#define BK      32
#define LDK     40        // LDS k-stride in halves: 80B rows -> b128-aligned,
                          // bank walk 20*m mod 64 is conflict-free for 16 lanes

// ---- CDNA5 async global->LDS copy (ASYNCcnt-tracked), via inline asm -------
// LDS operand is the per-lane LDS byte address: for a flat shared pointer the
// hardware uses addr[31:0] as the LDS address (ISA 10.2 aperture rules).
__device__ __forceinline__ void async_copy_b128(const _Float16* g, _Float16* l) {
  unsigned lds_addr = (unsigned)(uintptr_t)(void*)l;
  unsigned long long gaddr = (unsigned long long)(uintptr_t)(const void*)g;
  asm volatile("global_load_async_to_lds_b128 %0, %1, off"
               :
               : "v"(lds_addr), "v"(gaddr)
               : "memory");
}
__device__ __forceinline__ void wait_asynccnt0() {
  asm volatile("s_wait_asynccnt 0" ::: "memory");
}

// ---------------------------------------------------------------- encode ----
__global__ __launch_bounds__(256)
void encode_kernel(const float* __restrict__ pts,
                   const float* __restrict__ A,
                   const float* __restrict__ B,
                   _Float16* __restrict__ eA,
                   _Float16* __restrict__ eB) {
  const int i = blockIdx.x;
  const float x0 = pts[i * 3 + 0] * INV_R;
  const float x1 = pts[i * 3 + 1] * INV_R;
  const float x2 = pts[i * 3 + 2] * INV_R;
  for (int j = threadIdx.x; j < ENC + PP; j += 256) {
    if (j < ENC) {
      float ph = x0 * A[j] + x1 * A[ENC + j] + x2 * A[2 * ENC + j];
      float s, c;
      __sincosf(ph, &s, &c);
      eA[(size_t)i * EAW + j]       = (_Float16)c;
      eA[(size_t)i * EAW + ENC + j] = (_Float16)s;
    } else {
      const int jb = j - ENC;
      float ph = x0 * B[jb] + x1 * B[PP + jb] + x2 * B[2 * PP + jb];
      float s, c;
      __sincosf(ph, &s, &c);
      eB[(size_t)i * EBW + jb]      = (_Float16)c;
      eB[(size_t)i * EBW + PP + jb] = (_Float16)s;
    }
  }
}

// ---------------------------------------------------------------- helpers ---
union H4 { uint2 u; _Float16 h[4]; };          // 4x4 register-transpose piece
union FRAG { v16h v; uint4 q[2]; };

// A-fragment (16x32 f16, M x K) from LDS stored [m][k], stride LDK halves.
// ISA 7.12.2 A layout: VGPRs 0-3 hold K = klo..klo+7 (klo = 8*(lane>=16)),
// VGPRs 4-7 hold K = 16+klo..16+klo+7 -> two contiguous 16B runs.
__device__ __forceinline__ v16h load_a_frag(const _Float16* lds, int mBase, int lane) {
  FRAG f;
  const int m   = mBase + (lane & 15);
  const int klo = (lane >> 4) * 8;
  f.q[0] = *(const uint4*)&lds[m * LDK + klo];
  f.q[1] = *(const uint4*)&lds[m * LDK + 16 + klo];
  return f.v;
}

// B-fragment (32x16 f16, K x N) from LDS stored [n][k], stride LDK halves.
// Same K order as A: VGPRs 0-3 hold K = klo..klo+7 (klo = 8*(lane>=16)),
// VGPRs 4-7 hold K = 16+klo..16+klo+7 for column n=lane&15 -> two ds_load_b128.
__device__ __forceinline__ v16h load_b_frag(const _Float16* lds, int nBase, int lane) {
  FRAG f;
  const int n  = nBase + (lane & 15);
  const int k0 = (lane >> 4) * 8;
  f.q[0] = *(const uint4*)&lds[n * LDK + k0];
  f.q[1] = *(const uint4*)&lds[n * LDK + k0 + 16];
  return f.v;
}

// ---------------------------------------------------------------- gemm1 -----
// M[8192 x 1024] = (1/N) * eB^T @ eA       (f16 in, f32 acc, f16 out)
__global__ __launch_bounds__(256)
void gemm1_kernel(const _Float16* __restrict__ eBp,
                  const _Float16* __restrict__ eAp,
                  _Float16* __restrict__ Mout) {
  __shared__ alignas(16) _Float16 ldsA[128 * LDK];  // [m][k]
  __shared__ alignas(16) _Float16 ldsB[128 * LDK];  // [n][k]

  const int tid  = threadIdx.x;
  const int lane = tid & 31;
  const int wave = tid >> 5;
  const int wm   = wave >> 2;  // 0..1  (64 rows each)
  const int wn   = wave & 3;   // 0..3  (32 cols each)
  const int mBlock = blockIdx.y * 128;   // rows of M  (eB feature index)
  const int nBlock = blockIdx.x * 128;   // cols of M  (eA feature index)

  v8f acc[4][2];
#pragma unroll
  for (int mt = 0; mt < 4; ++mt)
#pragma unroll
    for (int nt = 0; nt < 2; ++nt)
      acc[mt][nt] = (v8f)(0.0f);

  const int cg = (tid & 31) * 4;   // column within block: 0,4,..,124
  const int kg = (tid >> 5) * 4;   // k within step:       0,4,..,28

  for (int k0 = 0; k0 < NPTS; k0 += BK) {
    const _Float16* gA = eBp + (size_t)(k0 + kg) * EBW + mBlock + cg;
    const _Float16* gB = eAp + (size_t)(k0 + kg) * EAW + nBlock + cg;
    H4 ra[4], rb[4];
#pragma unroll
    for (int r = 0; r < 4; ++r) {
      ra[r].u = *(const uint2*)(gA + (size_t)r * EBW);
      rb[r].u = *(const uint2*)(gB + (size_t)r * EAW);
    }
    __builtin_prefetch(gA + (size_t)BK * EBW, 0, 1);
    __builtin_prefetch(gB + (size_t)BK * EAW, 0, 1);

    __syncthreads();  // previous iteration's LDS reads done
#pragma unroll
    for (int j = 0; j < 4; ++j) {   // 4x4 register transpose, b64 stores
      H4 ca, cb;
#pragma unroll
      for (int r = 0; r < 4; ++r) { ca.h[r] = ra[r].h[j]; cb.h[r] = rb[r].h[j]; }
      *(uint2*)&ldsA[(cg + j) * LDK + kg] = ca.u;   // [m][k]
      *(uint2*)&ldsB[(cg + j) * LDK + kg] = cb.u;   // [n][k]
    }
    __syncthreads();  // LDS ready

    v16h bf0 = load_b_frag(ldsB, wn * 32 + 0,  lane);
    v16h bf1 = load_b_frag(ldsB, wn * 32 + 16, lane);
#pragma unroll
    for (int mt = 0; mt < 4; ++mt) {
      v16h af = load_a_frag(ldsA, wm * 64 + mt * 16, lane);
      acc[mt][0] = __builtin_amdgcn_wmma_f32_16x16x32_f16(
          false, af, false, bf0, (short)0, acc[mt][0], false, false);
      acc[mt][1] = __builtin_amdgcn_wmma_f32_16x16x32_f16(
          false, af, false, bf1, (short)0, acc[mt][1], false, false);
    }
  }

  // Epilogue: D layout: VGPR v = row M (+8 for upper lane half), N = lane&15.
  const float invN = 1.0f / (float)NPTS;
#pragma unroll
  for (int mt = 0; mt < 4; ++mt)
#pragma unroll
    for (int nt = 0; nt < 2; ++nt)
#pragma unroll
      for (int v = 0; v < 8; ++v) {
        const int gm = mBlock + wm * 64 + mt * 16 + (lane >> 4) * 8 + v;
        const int gn = nBlock + wn * 32 + nt * 16 + (lane & 15);
        Mout[(size_t)gm * EAW + gn] = (_Float16)(acc[mt][nt][v] * invN);
      }
}

// ---------------------------------------------------------------- gemm2 -----
// G[65536 x 1024] = eB @ M                 (f16 in, f32 out)
__global__ __launch_bounds__(256)
void gemm2_kernel(const _Float16* __restrict__ eBp,
                  const _Float16* __restrict__ Mm,
                  float* __restrict__ G) {
  __shared__ alignas(16) _Float16 ldsA[128 * LDK];  // [m][k]
  __shared__ alignas(16) _Float16 ldsB[128 * LDK];  // [n][k]

  const int tid  = threadIdx.x;
  const int lane = tid & 31;
  const int wave = tid >> 5;
  const int wm   = wave >> 2;
  const int wn   = wave & 3;
  const int mBlock = blockIdx.y * 128;   // point rows
  const int nBlock = blockIdx.x * 128;   // eA feature cols

  v8f acc[4][2];
#pragma unroll
  for (int mt = 0; mt < 4; ++mt)
#pragma unroll
    for (int nt = 0; nt < 2; ++nt)
      acc[mt][nt] = (v8f)(0.0f);

  // A fill: eB is row-major [point][k] == [m][k]: straight copy -> async DMA
  const int am = tid >> 1;          // 0..127
  const int ah = (tid & 1) * 16;    // k half-group
  // B fill: M[k][n] -> [n][k] via 4x4 register transpose
  const int cg = (tid & 31) * 4;
  const int kg = (tid >> 5) * 4;

  for (int k0 = 0; k0 < EBW; k0 += BK) {
    const _Float16* gA = eBp + (size_t)(mBlock + am) * EBW + k0 + ah;
    const _Float16* gB = Mm + (size_t)(k0 + kg) * EAW + nBlock + cg;
    H4 rb[4];
#pragma unroll
    for (int r = 0; r < 4; ++r)
      rb[r].u = *(const uint2*)(gB + (size_t)r * EAW);
    __builtin_prefetch(gA + BK, 0, 1);
    __builtin_prefetch(gB + (size_t)BK * EAW, 0, 1);

    __syncthreads();  // previous iteration's LDS reads done
    // Direct global->LDS DMA (no VGPR bounce), tracked by ASYNCcnt.
    async_copy_b128(gA,     &ldsA[am * LDK + ah]);
    async_copy_b128(gA + 8, &ldsA[am * LDK + ah + 8]);
#pragma unroll
    for (int j = 0; j < 4; ++j) {
      H4 cb;
#pragma unroll
      for (int r = 0; r < 4; ++r) cb.h[r] = rb[r].h[j];
      *(uint2*)&ldsB[(cg + j) * LDK + kg] = cb.u;
    }
    wait_asynccnt0();
    __syncthreads();  // LDS ready

    v16h bf0 = load_b_frag(ldsB, wn * 32 + 0,  lane);
    v16h bf1 = load_b_frag(ldsB, wn * 32 + 16, lane);
#pragma unroll
    for (int mt = 0; mt < 4; ++mt) {
      v16h af = load_a_frag(ldsA, wm * 64 + mt * 16, lane);
      acc[mt][0] = __builtin_amdgcn_wmma_f32_16x16x32_f16(
          false, af, false, bf0, (short)0, acc[mt][0], false, false);
      acc[mt][1] = __builtin_amdgcn_wmma_f32_16x16x32_f16(
          false, af, false, bf1, (short)0, acc[mt][1], false, false);
    }
  }

#pragma unroll
  for (int mt = 0; mt < 4; ++mt)
#pragma unroll
    for (int nt = 0; nt < 2; ++nt)
#pragma unroll
      for (int v = 0; v < 8; ++v) {
        const int gm = mBlock + wm * 64 + mt * 16 + (lane >> 4) * 8 + v;
        const int gn = nBlock + wn * 32 + nt * 16 + (lane & 15);
        G[(size_t)gm * EAW + gn] = acc[mt][nt][v];
      }
}

// -------------------------------------------------------------- finalize ----
__global__ __launch_bounds__(256)
void finalize_kernel(const _Float16* __restrict__ eA,
                     const float* __restrict__ G,
                     float* __restrict__ out) {
  __shared__ float red[256];
  const int i = blockIdx.x;
  const _Float16* ea = eA + (size_t)i * EAW;
  const float*    g  = G  + (size_t)i * EAW;

  float re[2], im[2];
  float local = 0.0f;
#pragma unroll
  for (int t = 0; t < 2; ++t) {
    const int j = threadIdx.x + t * 256;
    const float c  = (float)ea[j];
    const float s  = (float)ea[ENC + j];
    const float gc = g[j];
    const float gs = g[ENC + j];
    const float den = c * c + s * s;
    const float r = (gc * c + gs * s) / den;
    const float m = (gs * c - gc * s) / den;
    re[t] = r;
    im[t] = m;
    local += r * r + m * m;
  }
  red[threadIdx.x] = local;
  __syncthreads();
  for (int sft = 128; sft > 0; sft >>= 1) {
    if (threadIdx.x < sft) red[threadIdx.x] += red[threadIdx.x + sft];
    __syncthreads();
  }
  const float scale = SQRT_D / sqrtf(red[0]);
#pragma unroll
  for (int t = 0; t < 2; ++t) {
    const int j = threadIdx.x + t * 256;
    out[(size_t)i * EAW + j]       = re[t] * scale;
    out[(size_t)i * EAW + ENC + j] = im[t] * scale;
  }
}

// ---------------------------------------------------------------- launch ----
extern "C" void kernel_launch(void* const* d_in, const int* in_sizes, int n_in,
                              void* d_out, int out_size, void* d_ws, size_t ws_size,
                              hipStream_t stream) {
  const float* pts = (const float*)d_in[0];
  const float* A   = (const float*)d_in[1];
  const float* B   = (const float*)d_in[2];
  float* out = (float*)d_out;

  char* ws = (char*)d_ws;
  const size_t eA_bytes = (size_t)NPTS * EAW * sizeof(_Float16);   // 128 MiB
  const size_t eB_bytes = (size_t)NPTS * EBW * sizeof(_Float16);   // 1 GiB
  const size_t M_bytes  = (size_t)EBW * EAW * sizeof(_Float16);    // 16 MiB
  _Float16* eA = (_Float16*)ws;
  _Float16* eB = (_Float16*)(ws + eA_bytes);
  _Float16* Mm = (_Float16*)(ws + eA_bytes + eB_bytes);
  float*    G  = (float*)   (ws + eA_bytes + eB_bytes + M_bytes);  // 256 MiB

  encode_kernel<<<NPTS, 256, 0, stream>>>(pts, A, B, eA, eB);
  gemm1_kernel<<<dim3(EAW / 128, EBW / 128), 256, 0, stream>>>(eB, eA, Mm);
  gemm2_kernel<<<dim3(EAW / 128, NPTS / 128), 256, 0, stream>>>(eB, Mm, G);
  finalize_kernel<<<NPTS, 256, 0, stream>>>(eA, G, out);
}
